// BiSLSTM_25305947308037
// MI455X (gfx1250) — matrix operand order repaired and checked by reading, not compiled, not run on hardware
//
#include <hip/hip_runtime.h>
#include <hip/hip_bf16.h>
#include <math.h>

// ---------------------------------------------------------------------------
// BiSLSTM for MI455X (gfx1250).
//  - bf16 WMMA (v_wmma_f32_16x16x32_bf16) for all GEMMs, f32 accumulate.
//  - Time-parallel x@Wx hoisted into one big WMMA GEMM (Zx).
//  - Recurrence: 16 workgroups per direction, each owning a 16-column slice.
//    Weight slices live in LDS (80KB) for the whole scan; per-step cross-WG
//    h/s state exchange goes through L2 with monotonic atomic counters, and
//    the full state image is gathered back into LDS with the Tensor Data
//    Mover (tensor_load_to_lds + s_wait_tensorcnt).
// ---------------------------------------------------------------------------

typedef __bf16 v8bf  __attribute__((ext_vector_type(8)));
typedef __bf16 v16bf __attribute__((ext_vector_type(16)));
typedef float  v8f   __attribute__((ext_vector_type(8)));
typedef unsigned int u32x4 __attribute__((ext_vector_type(4)));
typedef unsigned int u32x8 __attribute__((ext_vector_type(8)));

union V16 { v16bf v; v8bf h[2]; };

#define Bn   32
#define Sn   512
#define En   256
#define Hn   256
#define H4   1024
#define NWG  16                            // recur workgroups per direction

// ws element counts / offsets (in elements of their own type)
#define XBF_ELEMS   (Bn*Sn*En)            // 4194304 bf16
#define WX_ELEMS    (En*H4)               // 262144
#define WH_ELEMS    (Hn*H4)
#define WS_ELEMS    (Hn*H4)
#define UH_ELEMS    (Hn*Hn)               // 65536
#define US_ELEMS    (Hn*Hn)
#define DIR_PK      (WX_ELEMS+WH_ELEMS+WS_ELEMS+UH_ELEMS+US_ELEMS) // 917504
#define PK_WX_OFF   0
#define PK_WH_OFF   (WX_ELEMS)
#define PK_WS_OFF   (WX_ELEMS+WH_ELEMS)
#define PK_UH_OFF   (WX_ELEMS+WH_ELEMS+WS_ELEMS)
#define PK_US_OFF   (WX_ELEMS+WH_ELEMS+WS_ELEMS+UH_ELEMS)
#define STATE_ELEMS (Bn*Hn)               // 8192 bf16 per state image

__device__ __forceinline__ float sigf(float x) {
    return 1.0f / (1.0f + __expf(-x));
}

// ---------------------------------------------------------------------------
__global__ void cvt_bf16_kernel(const float* __restrict__ src,
                                __bf16* __restrict__ dst, int n) {
    int i = blockIdx.x * blockDim.x + threadIdx.x;
    if (i < n) dst[i] = (__bf16)src[i];
}

__global__ void zero_u32_kernel(unsigned int* __restrict__ p, int n) {
    int i = blockIdx.x * blockDim.x + threadIdx.x;
    if (i < n) p[i] = 0u;
}

// ---------------------------------------------------------------------------
// Pack [K,N] row-major fp32 weight into WMMA B-fragment blobs:
//   dst[((nt*(K/32) + ks)*32 + lane)*16 + e] = W[ks*32 + ((e&8)<<1) + (lane>>4)*8 + (e&7)][nt*16 + (lane&15)]
// ---------------------------------------------------------------------------
__global__ void pack_b_kernel(const float* __restrict__ src,
                              __bf16* __restrict__ dst, int K, int N) {
    int tid = blockIdx.x * blockDim.x + threadIdx.x;
    int total = K * N;
    if (tid >= total) return;
    int KS   = K >> 5;
    int e    = tid & 15;
    int lane = (tid >> 4) & 31;
    int ks   = (tid >> 9) % KS;
    int nt   = (tid >> 9) / KS;
    dst[tid] = (__bf16)src[(size_t)(ks * 32 + ((e & 8) << 1) + ((lane >> 4) << 3) + (e & 7)) * N + nt * 16 + (lane & 15)];
}

__device__ __forceinline__ v16bf load_b_frag(const __bf16* pk, int nt, int ks,
                                             int KS, int lane) {
    const __bf16* p = pk + (((size_t)nt * KS + ks) * 32 + lane) * 16;
    V16 b;
    b.h[0] = *(const v8bf*)p;
    b.h[1] = *(const v8bf*)(p + 8);
    return b.v;
}

// B-fragment from an LDS blob laid out [.. fragment ..][32 lanes][16]
__device__ __forceinline__ v16bf load_b_lds(const __bf16* frag, int lane) {
    const __bf16* p = frag + lane * 16;
    V16 b;
    b.h[0] = *(const v8bf*)p;
    b.h[1] = *(const v8bf*)(p + 8);
    return b.v;
}

// A-fragment (16x32 bf16) from a row-major [rows][256] bf16 buffer.
// ISA 16-bit A layout: lanes 0-15 -> K {0..7,16..23}, lanes 16-31 -> K {8..15,24..31}.
__device__ __forceinline__ v16bf load_a_frag(const __bf16* base, int row,
                                             int kbase, int lane) {
    int koff = (lane < 16) ? 0 : 8;
    const __bf16* p = base + (size_t)row * 256 + kbase + koff;
    V16 a;
    a.h[0] = *(const v8bf*)p;
    a.h[1] = *(const v8bf*)(p + 16);
    return a.v;
}

// ---------------------------------------------------------------------------
// Zx GEMM: Zx[dir][t*32+b][n] = x[b,t,:] @ Wx_dir + b_dir[n]
// ---------------------------------------------------------------------------
__global__ __launch_bounds__(256) void zx_gemm_kernel(
    const __bf16* __restrict__ xbf,
    const __bf16* __restrict__ wxpk_f, const __bf16* __restrict__ wxpk_r,
    const float* __restrict__ b_f, const float* __restrict__ b_r,
    float* __restrict__ Zx) {
    int lane = threadIdx.x & 31;
    int wave = threadIdx.x >> 5;
    long tile = (long)blockIdx.x * 8 + wave;   // 131072 tiles
    int dir   = (int)(tile >> 16);
    int rem   = (int)(tile & 65535);
    int mtile = rem >> 6;
    int ntile = rem & 63;
    const __bf16* wpk  = dir ? wxpk_r : wxpk_f;
    const float*  bias = dir ? b_r : b_f;

    int rloc = lane & 15;
    int grow = mtile * 16 + rloc;               // flattened (t,b) row
    int t = grow >> 5, b = grow & 31;
    const __bf16* abase = xbf + ((size_t)b * Sn + t) * En;
    int koff = (lane < 16) ? 0 : 8;

    v8f acc = {};
#pragma unroll
    for (int ks = 0; ks < 8; ++ks) {
        V16 a;
        const __bf16* ap = abase + ks * 32 + koff;
        a.h[0] = *(const v8bf*)ap;
        a.h[1] = *(const v8bf*)(ap + 16);
        v16bf bb = load_b_frag(wpk, ntile, ks, 8, lane);
        acc = __builtin_amdgcn_wmma_f32_16x16x32_bf16(
            false, a.v, false, bb, (short)0, acc, false, false);
    }

    int n = ntile * 16 + rloc;
    float bv = bias[n];
    float* zcol = Zx + (size_t)dir * 16384 * H4 + n;
#pragma unroll
    for (int j = 0; j < 8; ++j) {
        int gr = mtile * 16 + j + ((lane >= 16) ? 8 : 0);
        zcol[(size_t)gr * H4] = acc[j] + bv;
    }
}

// ---------------------------------------------------------------------------
// TDM: gather one full 16KB state image (32x256 bf16) from global into LDS.
// D# per CDNA5 ISA ch.8: group0 = {flags, lds_addr, global_addr, type},
// group1 = 1-D tile, data_size=8B, tensor_dim0=tile_dim0=2048.
// ---------------------------------------------------------------------------
__device__ __forceinline__ void tdm_gather_state(const __bf16* gsrc,
                                                 __bf16* lds_dst) {
    unsigned int lds_off = (unsigned int)(size_t)lds_dst;   // low 32 = LDS offset
    unsigned long long ga = (unsigned long long)(size_t)gsrc;
    u32x4 g0 = { 1u,                                   // count=1, user mode
                 lds_off,
                 (unsigned int)ga,
                 (unsigned int)(ga >> 32) | 0x80000000u };  // type=2 ("image")
    u32x8 g1 = { 0x00030000u,   // data_size=3 (8B), no multicast/pad/iterate
                 0x08000000u,   // tensor_dim0 = 2048 (bits 79:48, low part)
                 0x00010000u,   // tensor_dim0 hi = 0, tensor_dim1 = 1
                 0x08000000u,   // tile_dim0 = 2048 (bits 127:112)
                 0x00000001u,   // tile_dim1 = 1
                 2048u,         // tensor_dim0_stride
                 0u, 0u };
    asm volatile("tensor_load_to_lds %0, %1" : : "s"(g0), "s"(g1) : "memory");
    __builtin_amdgcn_s_wait_tensorcnt(0);
}

// ---------------------------------------------------------------------------
// Recurrence. Grid = 32 WGs: blockIdx = dir*16 + w. WG (dir,w) owns h-columns
// [w*16, w*16+16). 2 waves per WG (mt = row-tile). Each wave keeps c, blended
// h, s for its 16x16 tile in registers across all 512 steps. Per-step
// cross-WG exchange: publish 16-col slice to hg/sg in L2, arrive on a
// monotonic counter, spin, then TDM-gather the full image back into LDS.
// ---------------------------------------------------------------------------
__global__ __launch_bounds__(64) void recur_kernel(
    const __bf16* __restrict__ pk,
    const float*  __restrict__ Zx,
    const float*  __restrict__ mask,
    const float*  __restrict__ bs_f, const float* __restrict__ bs_r,
    __bf16* __restrict__ hg,          // [2 dir][2 buf][32][256]
    __bf16* __restrict__ sg,          // [2 dir][2 buf][32][256]
    unsigned int* __restrict__ cnt,   // [4]: h-phase dir0/1, s-phase dir0/1
    float* __restrict__ out) {
    __shared__ __align__(16) __bf16 hbuf[2][STATE_ELEMS];   // 2 x 16KB
    __shared__ __align__(16) __bf16 sbuf[2][STATE_ELEMS];   // 2 x 16KB
    __shared__ __align__(16) __bf16 whl[4 * 8 * 512];       // 32KB [q][ks][512]
    __shared__ __align__(16) __bf16 wsl[4 * 8 * 512];       // 32KB
    __shared__ __align__(16) __bf16 uhl[8 * 512];           // 8KB  [ks][512]
    __shared__ __align__(16) __bf16 usl[8 * 512];           // 8KB

    const int dir  = blockIdx.x >> 4;
    const int w    = blockIdx.x & 15;
    const int lane = threadIdx.x & 31;
    const int mt   = threadIdx.x >> 5;     // wave id: row-tile 0..1

    const __bf16* whp = pk + (size_t)dir * DIR_PK + PK_WH_OFF;
    const __bf16* wsp = pk + (size_t)dir * DIR_PK + PK_WS_OFF;
    const __bf16* uhp = pk + (size_t)dir * DIR_PK + PK_UH_OFF;
    const __bf16* usp = pk + (size_t)dir * DIR_PK + PK_US_OFF;
    const float*  bsv = dir ? bs_r : bs_f;

    // ---- stage weight slices into LDS (once) ----
    for (int q = 0; q < 4; ++q) {
        const uint4* sh = (const uint4*)(whp + (size_t)(q * 16 + w) * 4096);
        const uint4* ss = (const uint4*)(wsp + (size_t)(q * 16 + w) * 4096);
        uint4* dh = (uint4*)(whl + q * 4096);
        uint4* ds = (uint4*)(wsl + q * 4096);
        for (int i = threadIdx.x; i < 512; i += 64) { dh[i] = sh[i]; ds[i] = ss[i]; }
    }
    {
        const uint4* su = (const uint4*)(uhp + (size_t)w * 4096);
        const uint4* sv = (const uint4*)(usp + (size_t)w * 4096);
        uint4* du = (uint4*)uhl;
        uint4* dv = (uint4*)usl;
        for (int i = threadIdx.x; i < 512; i += 64) { du[i] = su[i]; dv[i] = sv[i]; }
        uint4 z = {0u, 0u, 0u, 0u};
        uint4* h0 = (uint4*)&hbuf[0][0];
        uint4* s0 = (uint4*)&sbuf[0][0];
        for (int i = threadIdx.x; i < 1024; i += 64) { h0[i] = z; s0[i] = z; }
    }
    v8f c_prev = {}, h_prev = {}, s_prev = {};
    __syncthreads();

    const int rbase = mt * 16 + ((lane >= 16) ? 8 : 0);  // +j -> batch row
    const int ncol  = w * 16 + (lane & 15);              // h column
    const int arow  = mt * 16 + (lane & 15);             // A-fragment row
    unsigned int* cnt_h = cnt + dir;
    unsigned int* cnt_s = cnt + 2 + dir;
    float* out_h = out;
    float* out_c = out + (size_t)Sn * Bn * 512;
    float* out_s = out + (size_t)2 * Sn * Bn * 512;

    for (int tt = 0; tt < Sn; ++tt) {
        const int t   = dir ? (Sn - 1 - tt) : tt;
        const int cur = tt & 1, nxt = cur ^ 1;
        const unsigned int tgt = (unsigned int)(NWG * (tt + 1));

        float mv[8];
#pragma unroll
        for (int j = 0; j < 8; ++j) mv[j] = mask[(size_t)(rbase + j) * Sn + t];

        // ---- stage 1: z = [h;s] @ [Wh;Ws] + Zx[t]  (weights from LDS) ----
        v8f acc[4];
        const float* zb = Zx + ((size_t)dir * 16384 + (size_t)t * Bn) * H4;
#pragma unroll
        for (int q = 0; q < 4; ++q) {
            int n = q * Hn + ncol;
#pragma unroll
            for (int j = 0; j < 8; ++j)
                acc[q][j] = zb[(size_t)(rbase + j) * H4 + n];
        }
#pragma unroll
        for (int ks = 0; ks < 16; ++ks) {
            const __bf16* abuf = (ks < 8) ? &hbuf[cur][0] : &sbuf[cur][0];
            v16bf a = load_a_frag(abuf, arow, (ks & 7) * 32, lane);
            const __bf16* wl = (ks < 8) ? whl : wsl;
#pragma unroll
            for (int q = 0; q < 4; ++q) {
                v16bf bb = load_b_lds(wl + (size_t)(q * 8 + (ks & 7)) * 512, lane);
                acc[q] = __builtin_amdgcn_wmma_f32_16x16x32_bf16(
                    false, a, false, bb, (short)0, acc[q], false, false);
            }
        }
        // gates + state update; publish blended h slice to L2
        __bf16* hpub = hg + ((size_t)dir * 2 + nxt) * STATE_ELEMS;
#pragma unroll
        for (int j = 0; j < 8; ++j) {
            float iv = sigf(acc[0][j]);
            float fv = sigf(acc[1][j]);
            float gv = tanhf(acc[2][j]);
            float ov = sigf(acc[3][j]);
            float cn = fv * c_prev[j] + iv * gv;
            float hn = ov * tanhf(cn);
            float m  = mv[j];
            float cb = m * cn + (1.0f - m) * c_prev[j];
            float hb = m * hn + (1.0f - m) * h_prev[j];
            c_prev[j] = cb;
            h_prev[j] = hb;
            int brow = rbase + j;
            hpub[(size_t)brow * Hn + ncol] = (__bf16)hb;
            size_t o = ((size_t)t * Bn + brow) * 512 + dir * Hn + ncol;
            out_h[o] = hb;
            out_c[o] = cb;
        }
        // arrive + wait: all 16 WGs of this direction published h
        __threadfence();
        __syncthreads();
        if (threadIdx.x == 0) {
            atomicAdd(cnt_h, 1u);
            while (__hip_atomic_load(cnt_h, __ATOMIC_ACQUIRE,
                                     __HIP_MEMORY_SCOPE_AGENT) < tgt)
                __builtin_amdgcn_s_sleep(1);
            __threadfence();
        }
        __syncthreads();
        // TDM-gather full h image into LDS (wave 0 issues; TENSORcnt tracks)
        if (mt == 0)
            tdm_gather_state(hg + ((size_t)dir * 2 + nxt) * STATE_ELEMS,
                             &hbuf[nxt][0]);
        __syncthreads();

        // ---- stage 2: s_new = tanh([h_new;s] @ [Uh;Us] + bs) ----
        float bsc = bsv[ncol];
        v8f sa;
#pragma unroll
        for (int j = 0; j < 8; ++j) sa[j] = bsc;
#pragma unroll
        for (int ks = 0; ks < 16; ++ks) {
            const __bf16* abuf = (ks < 8) ? &hbuf[nxt][0] : &sbuf[cur][0];
            v16bf a = load_a_frag(abuf, arow, (ks & 7) * 32, lane);
            const __bf16* ul = (ks < 8) ? uhl : usl;
            v16bf bb = load_b_lds(ul + (size_t)(ks & 7) * 512, lane);
            sa = __builtin_amdgcn_wmma_f32_16x16x32_bf16(
                false, a, false, bb, (short)0, sa, false, false);
        }
        __bf16* spub = sg + ((size_t)dir * 2 + nxt) * STATE_ELEMS;
#pragma unroll
        for (int j = 0; j < 8; ++j) {
            float sn = tanhf(sa[j]);
            float m  = mv[j];
            float sb = m * sn + (1.0f - m) * s_prev[j];
            s_prev[j] = sb;
            int brow = rbase + j;
            spub[(size_t)brow * Hn + ncol] = (__bf16)sb;
            out_s[((size_t)t * Bn + brow) * 512 + dir * Hn + ncol] = sb;
        }
        __threadfence();
        __syncthreads();
        if (threadIdx.x == 0) {
            atomicAdd(cnt_s, 1u);
            while (__hip_atomic_load(cnt_s, __ATOMIC_ACQUIRE,
                                     __HIP_MEMORY_SCOPE_AGENT) < tgt)
                __builtin_amdgcn_s_sleep(1);
            __threadfence();
        }
        __syncthreads();
        if (mt == 0)
            tdm_gather_state(sg + ((size_t)dir * 2 + nxt) * STATE_ELEMS,
                             &sbuf[nxt][0]);
        __syncthreads();
    }
}

// ---------------------------------------------------------------------------
// Launcher
// ---------------------------------------------------------------------------
extern "C" void kernel_launch(void* const* d_in, const int* in_sizes, int n_in,
                              void* d_out, int out_size, void* d_ws, size_t ws_size,
                              hipStream_t stream) {
    const float* inputs = (const float*)d_in[0];
    const float* mask   = (const float*)d_in[1];
    const float* Wx_f = (const float*)d_in[3];
    const float* Wh_f = (const float*)d_in[4];
    const float* Ws_f = (const float*)d_in[5];
    const float* b_f  = (const float*)d_in[6];
    const float* Us_f = (const float*)d_in[7];
    const float* Uh_f = (const float*)d_in[8];
    const float* bs_f = (const float*)d_in[9];
    const float* Wx_r = (const float*)d_in[10];
    const float* Wh_r = (const float*)d_in[11];
    const float* Ws_r = (const float*)d_in[12];
    const float* b_r  = (const float*)d_in[13];
    const float* Us_r = (const float*)d_in[14];
    const float* Uh_r = (const float*)d_in[15];
    const float* bs_r = (const float*)d_in[16];

    // workspace layout
    __bf16* xbf = (__bf16*)d_ws;                           // 8 MB
    __bf16* pk  = xbf + XBF_ELEMS;                         // 3.5 MB
    float*  Zx  = (float*)(pk + 2 * (size_t)DIR_PK);       // 128 MB
    __bf16* hg  = (__bf16*)(Zx + (size_t)2 * 16384 * H4);  // 64 KB
    __bf16* sg  = hg + 4 * STATE_ELEMS;                    // 64 KB
    unsigned int* cnt = (unsigned int*)(sg + 4 * STATE_ELEMS);

    // 0) zero state images + sync counters (part of the captured graph)
    zero_u32_kernel<<<(32772 + 255) / 256, 256, 0, stream>>>(
        (unsigned int*)hg, 32772);

    // 1) x -> bf16
    cvt_bf16_kernel<<<XBF_ELEMS / 256, 256, 0, stream>>>(inputs, xbf, XBF_ELEMS);

    // 2) pack weights into WMMA B-fragment layout
    struct { const float* src; __bf16* dst; int K, N; } pkjobs[10] = {
        { Wx_f, pk + 0 * DIR_PK + PK_WX_OFF, En, H4 },
        { Wh_f, pk + 0 * DIR_PK + PK_WH_OFF, Hn, H4 },
        { Ws_f, pk + 0 * DIR_PK + PK_WS_OFF, Hn, H4 },
        { Uh_f, pk + 0 * DIR_PK + PK_UH_OFF, Hn, Hn },
        { Us_f, pk + 0 * DIR_PK + PK_US_OFF, Hn, Hn },
        { Wx_r, pk + 1 * DIR_PK + PK_WX_OFF, En, H4 },
        { Wh_r, pk + 1 * DIR_PK + PK_WH_OFF, Hn, H4 },
        { Ws_r, pk + 1 * DIR_PK + PK_WS_OFF, Hn, H4 },
        { Uh_r, pk + 1 * DIR_PK + PK_UH_OFF, Hn, Hn },
        { Us_r, pk + 1 * DIR_PK + PK_US_OFF, Hn, Hn },
    };
    for (int i = 0; i < 10; ++i) {
        int total = pkjobs[i].K * pkjobs[i].N;
        pack_b_kernel<<<(total + 255) / 256, 256, 0, stream>>>(
            pkjobs[i].src, pkjobs[i].dst, pkjobs[i].K, pkjobs[i].N);
    }

    // 3) Zx = x @ Wx + b, both directions
    zx_gemm_kernel<<<16384, 256, 0, stream>>>(
        xbf, pk + 0 * DIR_PK + PK_WX_OFF, pk + 1 * DIR_PK + PK_WX_OFF,
        b_f, b_r, Zx);

    // 4) recurrence: 16 WGs per direction, LDS-resident weights, TDM gathers
    recur_kernel<<<32, 64, 0, stream>>>(pk, Zx, mask, bs_f, bs_r,
                                        hg, sg, cnt, (float*)d_out);
}
